// FullyAttention_11699490915080
// MI455X (gfx1250) — compile-verified, matrix operand order checked by reading, not run on hardware
//
#include <hip/hip_runtime.h>
#include <hip/hip_bf16.h>

typedef __attribute__((ext_vector_type(16))) _Float16 v16h;
typedef __attribute__((ext_vector_type(8)))  _Float16 v8h;
typedef __attribute__((ext_vector_type(4)))  _Float16 v4h;
typedef __attribute__((ext_vector_type(8)))  float    v8f;
typedef __attribute__((ext_vector_type(4)))  float    v4f;

__device__ __forceinline__ v8f zero8() {
  v8f z;
#pragma unroll
  for (int i = 0; i < 8; ++i) z[i] = 0.0f;
  return z;
}

// 16-bit A/B fragment: lane holds one row (A) / col (B); two contiguous
// 8-element K-runs: [k0+koff .. +7] and [k0+16+koff .. +7], koff = (lane>>4)*8.
__device__ __forceinline__ v16h frag_f16(const _Float16* p) {
  v8h lo = *(const v8h*)(p);
  v8h hi = *(const v8h*)(p + 16);
  v16h r;
#pragma unroll
  for (int i = 0; i < 8; ++i) { r[i] = lo[i]; r[i + 8] = hi[i]; }
  return r;
}

// Same fragment but source is f32: vector-load, convert in-register.
__device__ __forceinline__ v16h frag_f32(const float* p) {
  v4f a0 = *(const v4f*)(p);
  v4f a1 = *(const v4f*)(p + 4);
  v4f b0 = *(const v4f*)(p + 16);
  v4f b1 = *(const v4f*)(p + 20);
  v16h r;
#pragma unroll
  for (int i = 0; i < 4; ++i) {
    r[i]      = (_Float16)a0[i];
    r[i + 4]  = (_Float16)a1[i];
    r[i + 8]  = (_Float16)b0[i];
    r[i + 12] = (_Float16)b1[i];
  }
  return r;
}

__device__ __forceinline__ v8f wmma32(v16h a, v16h b, v8f c) {
  return __builtin_amdgcn_wmma_f32_16x16x32_f16(false, a, false, b, (short)0, c,
                                                false, false);
}

// ---------------------------------------------------------------------------
// Kernel 0a: Uw16 = (f16) U_w   (256x1024 = 256K elements, one-shot)
// ---------------------------------------------------------------------------
__global__ void __launch_bounds__(256)
fa_convert_uw(const float* __restrict__ Uw, _Float16* __restrict__ Uw16) {
  const int i = (blockIdx.x * 256 + threadIdx.x) * 4;
  v4f v = *(const v4f*)(Uw + i);
  v4h o;
#pragma unroll
  for (int k = 0; k < 4; ++k) o[k] = (_Float16)v[k];
  *(v4h*)(Uw16 + i) = o;
}

// ---------------------------------------------------------------------------
// Kernel 0b: ctxT[b][h][c] = (f16) context[b][c][h]   (32x32 LDS tile)
// ---------------------------------------------------------------------------
__global__ void __launch_bounds__(256)
fa_transpose_ctx(const float* __restrict__ ctx, _Float16* __restrict__ ctxT) {
  __shared__ _Float16 tile[32][33];
  const int b  = blockIdx.z;
  const int c0 = blockIdx.x * 32;
  const int h0 = blockIdx.y * 32;
  const int x  = threadIdx.x;            // 0..31
  const size_t base = (size_t)b * (1024u * 1024u);
#pragma unroll
  for (int i = 0; i < 4; ++i) {
    int cl = threadIdx.y + i * 8;
    tile[cl][x] = (_Float16)ctx[base + (size_t)(c0 + cl) * 1024 + h0 + x];
  }
  __syncthreads();
#pragma unroll
  for (int i = 0; i < 4; ++i) {
    int hl = threadIdx.y + i * 8;
    ctxT[base + (size_t)(h0 + hl) * 1024 + c0 + x] = tile[x][hl];
  }
}

// ---------------------------------------------------------------------------
// Kernel 1: F = relu(X @ U_w^T + b) [* D]  -> f16,  M=16384, N=256, K=1024
// One wave computes a full 16x256 row-tile (16 accumulators): X streamed once.
// B fragments come from the pre-converted f16 Uw16 (pure b128 loads, no cvt).
// ---------------------------------------------------------------------------
__global__ void __launch_bounds__(256)
fa_feat_gemm(const float* __restrict__ X, const _Float16* __restrict__ Uw16,
             const float* __restrict__ Ub, const float* __restrict__ Dv,
             _Float16* __restrict__ F, int mulD) {
  const int lane = threadIdx.x & 31;
  const int wave = (blockIdx.x * blockDim.x + threadIdx.x) >> 5;  // 0..1023
  const int ncol = lane & 15;
  const int koff = ((lane >> 4) & 1) * 8;
  const int row  = wave * 16 + ncol;     // A-matrix row for this lane

  v8f acc[16];
#pragma unroll
  for (int j = 0; j < 16; ++j) acc[j] = zero8();

  const float* pa = X + (size_t)row * 1024 + koff;
  for (int kt = 0; kt < 32; ++kt) {
    const int k0 = kt * 32;
    __builtin_prefetch(pa + k0 + 64, 0, 1);
    v16h a = frag_f32(pa + k0);
#pragma unroll
    for (int j = 0; j < 16; ++j) {
      const _Float16* pb = Uw16 + (size_t)(j * 16 + ncol) * 1024 + k0 + koff;
      v16h bfrag = frag_f16(pb);
      acc[j] = wmma32(a, bfrag, acc[j]);
    }
  }

  const int mbase = (lane >> 4) * 8;     // C/D: m = vgpr + 8*(lane>>4)
#pragma unroll
  for (int j = 0; j < 16; ++j) {
    const int n = j * 16 + ncol;
    const float bias = Ub[n];
    const float dmul = mulD ? Dv[n] : 1.0f;
#pragma unroll
    for (int r = 0; r < 8; ++r) {
      float v = acc[j][r] + bias;
      v = v > 0.0f ? v : 0.0f;
      v *= dmul;
      F[(size_t)(wave * 16 + mbase + r) * 256 + n] = (_Float16)v;
    }
  }
}

// ---------------------------------------------------------------------------
// Kernel 2 (fused flash attention): for one (batch, 16-row query block):
//   loop over context in 128-column chunks:
//     A) score tile 16x128 via WMMA (each wave one 16x16 tile) -> LDS
//     B) online softmax update (running max/sum, rescale factors) -> P in LDS
//     C) rescale accumulators, mix-accumulate P @ ctxT chunk via WMMA
//   epilogue: divide by running sum, store f32 output.
// Block = 256 threads (8 waves); wave w owns output columns [128w, 128w+128).
// ---------------------------------------------------------------------------
__global__ void __launch_bounds__(256)
fa_fused_attn(const _Float16* __restrict__ OF, const _Float16* __restrict__ CF,
              const _Float16* __restrict__ CT, float* __restrict__ OUT) {
  __shared__ alignas(16) float    Sld[16][132];   // score chunk (padded)
  __shared__ alignas(16) _Float16 Pld[16][128];   // prob chunk (f16)
  __shared__ float mrow[16], lrow[16], facs[16];

  const int tid   = threadIdx.x;
  const int lane  = tid & 31;
  const int wv    = tid >> 5;          // 0..7
  const int blk   = blockIdx.x;        // 0..1023
  const int b     = blk >> 6;          // batch
  const int qb    = blk & 63;          // query block (16 rows)
  const int ncol  = lane & 15;
  const int half  = (lane >> 4) & 1;
  const int koff  = half * 8;
  const int mbase = half * 8;

  const _Float16* ofb = OF + (size_t)b * (1024u * 256u);
  const _Float16* cfb = CF + (size_t)b * (1024u * 256u);
  const _Float16* ctb = CT + (size_t)b * (1024u * 1024u);

  // Preload score A-fragments: o_feat rows qb*16..+15, full K=256 (64 VGPRs).
  v16h af[8];
  {
    const _Float16* pa = ofb + (size_t)(qb * 16 + ncol) * 256 + koff;
#pragma unroll
    for (int kt = 0; kt < 8; ++kt) af[kt] = frag_f16(pa + kt * 32);
  }

  if (tid < 16) { mrow[tid] = -3.0e38f; lrow[tid] = 0.0f; }
  v8f acc[8];
#pragma unroll
  for (int j = 0; j < 8; ++j) acc[j] = zero8();
  __syncthreads();

  const int row16 = tid >> 4;          // phase-B row (0..15)
  const int sub   = tid & 15;          // 16 threads per row, 8 cols each

  for (int c0 = 0; c0 < 1024; c0 += 128) {
    // ---- Phase A: score tile, wave wv covers cols [c0+16wv, +16) ----
    {
      v8f sc = zero8();
      const _Float16* pb = cfb + (size_t)(c0 + wv * 16 + ncol) * 256 + koff;
#pragma unroll
      for (int kt = 0; kt < 8; ++kt) {
        v16h bf = frag_f16(pb + kt * 32);
        sc = wmma32(af[kt], bf, sc);
      }
#pragma unroll
      for (int r = 0; r < 8; ++r) Sld[mbase + r][wv * 16 + ncol] = sc[r];
    }
    __syncthreads();

    // ---- Phase B: online softmax on the 16x128 chunk ----
    {
      float v[8];
#pragma unroll
      for (int i = 0; i < 8; ++i) v[i] = Sld[row16][sub * 8 + i];
      float cmax = v[0];
#pragma unroll
      for (int i = 1; i < 8; ++i) cmax = fmaxf(cmax, v[i]);
#pragma unroll
      for (int off = 8; off > 0; off >>= 1)
        cmax = fmaxf(cmax, __shfl_xor(cmax, off, 16));
      const float mold = mrow[row16];
      const float mnew = fmaxf(mold, cmax);
      float sum = 0.0f;
#pragma unroll
      for (int i = 0; i < 8; ++i) {
        float e = __expf(v[i] - mnew);
        sum += e;
        Pld[row16][sub * 8 + i] = (_Float16)e;
      }
#pragma unroll
      for (int off = 8; off > 0; off >>= 1) sum += __shfl_xor(sum, off, 16);
      if (sub == 0) {
        const float f = __expf(mold - mnew);
        lrow[row16] = lrow[row16] * f + sum;
        mrow[row16] = mnew;
        facs[row16] = f;
      }
    }
    __syncthreads();

    // ---- Phase C: rescale accumulators, then mix-accumulate (K=128) ----
    {
      float fr[8];
#pragma unroll
      for (int r = 0; r < 8; ++r) fr[r] = facs[mbase + r];
#pragma unroll
      for (int j = 0; j < 8; ++j)
#pragma unroll
        for (int r = 0; r < 8; ++r) acc[j][r] *= fr[r];

#pragma unroll
      for (int kt = 0; kt < 4; ++kt) {
        const int k0 = kt * 32;
        v16h a = frag_f16(&Pld[ncol][k0 + koff]);          // LDS A-fragment
#pragma unroll
        for (int j = 0; j < 8; ++j) {
          const int h = wv * 128 + j * 16 + ncol;
          v16h bf = frag_f16(ctb + (size_t)h * 1024 + c0 + k0 + koff);
          acc[j] = wmma32(a, bf, acc[j]);
        }
      }
    }
    __syncthreads();   // protect Pld/facs before next chunk's phase B
  }

  // ---- Epilogue: normalize by running sum and store ----
  float li[8];
#pragma unroll
  for (int r = 0; r < 8; ++r) li[r] = 1.0f / lrow[mbase + r];
  float* ob = OUT + (size_t)b * (1024u * 1024u) + (size_t)(qb * 16) * 1024;
#pragma unroll
  for (int j = 0; j < 8; ++j)
#pragma unroll
    for (int r = 0; r < 8; ++r)
      ob[(size_t)(mbase + r) * 1024 + wv * 128 + j * 16 + ncol] =
          acc[j][r] * li[r];
}

// ---------------------------------------------------------------------------
extern "C" void kernel_launch(void* const* d_in, const int* in_sizes, int n_in,
                              void* d_out, int out_size, void* d_ws,
                              size_t ws_size, hipStream_t stream) {
  const float* h_output  = (const float*)d_in[0];  // (16,1024,1024)
  const float* h_context = (const float*)d_in[1];  // (16,1024,1024)
  const float* context   = (const float*)d_in[2];  // (16,1024,1024)
  const float* U_w       = (const float*)d_in[3];  // (256,1024)
  const float* U_b       = (const float*)d_in[4];  // (256,)
  const float* Dv        = (const float*)d_in[5];  // (256,)
  float* out = (float*)d_out;

  char* ws = (char*)d_ws;
  _Float16* o_feat = (_Float16*)(ws);                           //  8 MB
  _Float16* c_feat = (_Float16*)(ws + ((size_t)8  << 20));      //  8 MB
  _Float16* ctxT   = (_Float16*)(ws + ((size_t)16 << 20));      // 32 MB
  _Float16* Uw16   = (_Float16*)(ws + ((size_t)48 << 20));      // 512 KB

  // 0) one-shot conversions: U_w -> f16; context -> transposed f16
  fa_convert_uw<<<256, 256, 0, stream>>>(U_w, Uw16);
  fa_transpose_ctx<<<dim3(32, 32, 16), dim3(32, 8), 0, stream>>>(context, ctxT);

  // 1) feature maps (M=16384 rows => 1024 wave-tiles => 128 blocks of 8 waves)
  fa_feat_gemm<<<128, 256, 0, stream>>>(h_output, Uw16, U_b, Dv, o_feat, 1);
  fa_feat_gemm<<<128, 256, 0, stream>>>(h_context, Uw16, U_b, Dv, c_feat, 0);

  // 2) fused score + softmax + mix: 16 batches x 64 query blocks
  fa_fused_attn<<<1024, 256, 0, stream>>>(o_feat, c_feat, ctxT, out);
}
